// MultiHeadSelfAttention_19215683682712
// MI455X (gfx1250) — matrix operand order repaired and checked by reading, compiled, not run on hardware
//
#include <hip/hip_runtime.h>
#include <hip/hip_bf16.h>

// MI455X / gfx1250 multi-head self-attention, f16 WMMA path, f32 accumulate.
// B=2, S=2048, E=1024, H=16, D=64.

#define B_ 2
#define S_ 2048
#define E_ 1024
#define H_ 16
#define D_ 64

typedef __attribute__((ext_vector_type(16))) _Float16 v16h;
typedef __attribute__((ext_vector_type(8)))  _Float16 v8h;
typedef __attribute__((ext_vector_type(8)))  float    v8f;

union Frag {
    v16h v;
    v8h  h[2];
};

__device__ __forceinline__ v8f wmma_f16(v16h a, v16h b, v8f c) {
    return __builtin_amdgcn_wmma_f32_16x16x32_f16(
        /*neg_a=*/false, a, /*neg_b=*/false, b,
        /*c_mod=*/(short)0, c, /*reuse_a=*/false, /*reuse_b=*/false);
}

// ---------------------------------------------------------------------------
// Stage 0: f32 -> f16 conversion (grid-stride).
// ---------------------------------------------------------------------------
__global__ void f32_to_f16_kernel(const float* __restrict__ in,
                                  _Float16* __restrict__ out, long n) {
    long i = (long)blockIdx.x * blockDim.x + threadIdx.x;
    long stride = (long)gridDim.x * blockDim.x;
    for (; i < n; i += stride) out[i] = (_Float16)in[i];
}

// ---------------------------------------------------------------------------
// WMMA GEMM: C[m,n] = sum_k A[m,k]*W[n,k] + bias[n]   (A MxK, W NxK, row-major)
// One wave computes a 32x64 tile (2x4 grid of 16x16 WMMA tiles): each k-step
// issues 12 b128 fragment loads for 8 WMMAs (1.5 loads/WMMA vs 4.0 for a
// 16x16 tile) without spilling (8 v8f accumulators = 64 VGPRs).
// Per-lane addresses are 32-bit offsets from the uniform base pointer so the
// backend can use SGPR-base + VGPR-offset addressing.
// mode 0: QKV epilogue -> Qh (scaled 1/8), Kh [B,H,S,D], Vt [B,H,D,S] (f16)
// mode 1: proj epilogue -> f32 out[m*N + n]
// ---------------------------------------------------------------------------
__global__ __launch_bounds__(256)
void gemm_wmma_kernel(const _Float16* __restrict__ A,
                      const _Float16* __restrict__ W,
                      const float* __restrict__ bias,
                      int M, int N, int K, int mode,
                      _Float16* __restrict__ Qh,
                      _Float16* __restrict__ Kh,
                      _Float16* __restrict__ Vt,
                      float* __restrict__ Out) {
    const int wave = threadIdx.x >> 5;
    const int lane = threadIdx.x & 31;
    const int la = lane & 15;   // row (A) / col (B,C)
    const int lb = lane >> 4;   // K-half selector / row-half (C)

    const long wtile = (long)blockIdx.x * 8 + wave;
    const int tilesN = N >> 6;                 // 64-wide wave tiles
    const int tm = (int)(wtile / tilesN);
    const int tn = (int)(wtile % tilesN);
    if (tm >= (M >> 5)) return;
    const int m0 = tm << 5;                    // 32 rows
    const int n0 = tn << 6;                    // 64 cols

    // A fragment: lane L -> row (L&15), K = kk + (L>>4)*16 + {0..15}
    unsigned aoff[2];
    // B fragment: lane L -> col (L&15), K = kk + (L>>4)*16 + {0..15}
    unsigned boff[4];
#pragma unroll
    for (int r = 0; r < 2; ++r)
        aoff[r] = (unsigned)((m0 + 16 * r + la) * K + lb * 16);
#pragma unroll
    for (int c = 0; c < 4; ++c)
        boff[c] = (unsigned)((n0 + 16 * c + la) * K + lb * 16);

    v8f acc[2][4] = {};

#pragma unroll 1
    for (int kk = 0; kk < K; kk += 32) {
        Frag af[2], bf[4];
#pragma unroll
        for (int r = 0; r < 2; ++r) {
            af[r].h[0] = *(const v8h*)(A + aoff[r] + kk);
            af[r].h[1] = *(const v8h*)(A + aoff[r] + kk + 8);
        }
#pragma unroll
        for (int c = 0; c < 4; ++c) {
            bf[c].h[0] = *(const v8h*)(W + boff[c] + kk);
            bf[c].h[1] = *(const v8h*)(W + boff[c] + kk + 8);
        }
        // Prefetch a few k-steps ahead (lowers to global_prefetch_b8).
        __builtin_prefetch(A + aoff[0] + kk + 256, 0, 0);
        __builtin_prefetch(W + boff[0] + kk + 256, 0, 0);
#pragma unroll
        for (int r = 0; r < 2; ++r)
#pragma unroll
            for (int c = 0; c < 4; ++c)
                acc[r][c] = wmma_f16(af[r].v, bf[c].v, acc[r][c]);
    }

    if (mode == 0) {
        // n0 is 64-aligned -> whole wave tile is one (t, head) slice.
        const int t  = n0 >> 10;          // 0=q, 1=k, 2=v
        const int hh = (n0 >> 6) & 15;    // head
#pragma unroll
        for (int c = 0; c < 4; ++c) {
            const int col = n0 + 16 * c + la;
            const int d   = 16 * c + la;
            const float bv = bias[col];
#pragma unroll
            for (int r = 0; r < 2; ++r) {
#pragma unroll
                for (int i = 0; i < 8; ++i) {
                    const int row = m0 + 16 * r + i + lb * 8;
                    const float v = acc[r][c][i] + bv;
                    const int bidx = row >> 11;
                    const int s = row & 2047;
                    if (t == 0) {
                        Qh[(((long)(bidx * H_ + hh)) * S_ + s) * D_ + d] =
                            (_Float16)(v * 0.125f);   // pre-apply 1/sqrt(D)
                    } else if (t == 1) {
                        Kh[(((long)(bidx * H_ + hh)) * S_ + s) * D_ + d] = (_Float16)v;
                    } else {
                        Vt[(((long)(bidx * H_ + hh)) * D_ + d) * S_ + s] = (_Float16)v;
                    }
                }
            }
        }
    } else {
#pragma unroll
        for (int c = 0; c < 4; ++c) {
            const int col = n0 + 16 * c + la;
            const float bv = bias[col];
#pragma unroll
            for (int r = 0; r < 2; ++r) {
#pragma unroll
                for (int i = 0; i < 8; ++i) {
                    const int row = m0 + 16 * r + i + lb * 8;
                    Out[(long)row * N + col] = acc[r][c][i] + bv;
                }
            }
        }
    }
}

// ---------------------------------------------------------------------------
// Flash attention: one wave per 16-query tile, streaming softmax over 32-key
// chunks. Scores = Q*K^T (Q pre-scaled), P converted to f16 A-fragments via a
// wave-private LDS transpose, O accumulated with WMMA. Output f16 row-major
// Ah[b][s][h*D+d] for the final projection GEMM.
// ---------------------------------------------------------------------------
__global__ __launch_bounds__(256)
void attn_kernel(const _Float16* __restrict__ Qh,
                 const _Float16* __restrict__ Kh,
                 const _Float16* __restrict__ Vt,
                 const int* __restrict__ mask,
                 _Float16* __restrict__ Ah) {
    __shared__ __align__(16) _Float16 ldsP[8][16 * 32];   // per-wave P tile

    const int wave = threadIdx.x >> 5;
    const int lane = threadIdx.x & 31;
    const int la = lane & 15;
    const int lb = lane >> 4;

    const long tile = (long)blockIdx.x * 8 + wave;  // B*H*(S/16) = 4096 tiles
    const int qt = (int)(tile & (S_ / 16 - 1));     // 0..127
    const int bh = (int)(tile >> 7);                // 0..31
    const int b  = bh >> 4;
    const int hh = bh & 15;
    const int q0 = qt << 4;

    const _Float16* Qbase = Qh + (long)bh * S_ * D_;
    const _Float16* Kbase = Kh + (long)bh * S_ * D_;
    const _Float16* Vbase = Vt + (long)bh * D_ * S_;
    const int* mrow = mask + (long)b * S_;

    // Q A-fragments for d-chunks 0..31 and 32..63.
    Frag aq0, aq1;
    {
        const _Float16* qrow = Qbase + (long)(q0 + la) * D_ + lb * 16;
        aq0.h[0] = *(const v8h*)(qrow);
        aq0.h[1] = *(const v8h*)(qrow + 8);
        aq1.h[0] = *(const v8h*)(qrow + 32);
        aq1.h[1] = *(const v8h*)(qrow + 40);
    }

    float mrun[8], lrun[8];
    v8f O[4] = {};
#pragma unroll
    for (int i = 0; i < 8; ++i) { mrun[i] = -1e30f; lrun[i] = 0.f; }

#pragma unroll 1
    for (int kc = 0; kc < S_; kc += 32) {
        // Score tiles: keys [kc, kc+16) and [kc+16, kc+32).
        v8f s0 = {}, s1 = {};
        {
            const _Float16* krow = Kbase + (long)(kc + la) * D_ + lb * 16;
            Frag b0, b1;
            b0.h[0] = *(const v8h*)(krow);      b0.h[1] = *(const v8h*)(krow + 8);
            b1.h[0] = *(const v8h*)(krow + 32); b1.h[1] = *(const v8h*)(krow + 40);
            s0 = wmma_f16(aq0.v, b0.v, s0);
            s0 = wmma_f16(aq1.v, b1.v, s0);
        }
        {
            const _Float16* krow = Kbase + (long)(kc + 16 + la) * D_ + lb * 16;
            Frag b0, b1;
            b0.h[0] = *(const v8h*)(krow);      b0.h[1] = *(const v8h*)(krow + 8);
            b1.h[0] = *(const v8h*)(krow + 32); b1.h[1] = *(const v8h*)(krow + 40);
            s1 = wmma_f16(aq0.v, b0.v, s1);
            s1 = wmma_f16(aq1.v, b1.v, s1);
        }

        const int mk0 = mrow[kc + la];
        const int mk1 = mrow[kc + 16 + la];

        // Online softmax update (rows i + 8*lb; columns spread over 16 lanes).
#pragma unroll
        for (int i = 0; i < 8; ++i) {
            float x0 = mk0 ? s0[i] : -1e30f;
            float x1 = mk1 ? s1[i] : -1e30f;
            float tmax = fmaxf(x0, x1);
            tmax = fmaxf(tmax, __shfl_xor(tmax, 1));
            tmax = fmaxf(tmax, __shfl_xor(tmax, 2));
            tmax = fmaxf(tmax, __shfl_xor(tmax, 4));
            tmax = fmaxf(tmax, __shfl_xor(tmax, 8));
            const float mnew = fmaxf(mrun[i], tmax);
            const float corr = __expf(mrun[i] - mnew);
            mrun[i] = mnew;
            const float p0 = mk0 ? __expf(x0 - mnew) : 0.f;
            const float p1 = mk1 ? __expf(x1 - mnew) : 0.f;
            float ps = p0 + p1;
            ps += __shfl_xor(ps, 1);
            ps += __shfl_xor(ps, 2);
            ps += __shfl_xor(ps, 4);
            ps += __shfl_xor(ps, 8);
            lrun[i] = lrun[i] * corr + ps;
            O[0][i] *= corr; O[1][i] *= corr; O[2][i] *= corr; O[3][i] *= corr;
            const int r = i + lb * 8;
            ldsP[wave][r * 32 + la]      = (_Float16)p0;
            ldsP[wave][r * 32 + 16 + la] = (_Float16)p1;
        }

        // Cross-lane LDS RAW: wait for the DS stores before re-reading.
        asm volatile("s_wait_dscnt 0x0" ::: "memory");

        Frag ap;
        {
            const _Float16* prow = &ldsP[wave][la * 32 + lb * 16];
            ap.h[0] = *(const v8h*)(prow);
            ap.h[1] = *(const v8h*)(prow + 8);
        }

        // O[:, 16j..16j+15] += P (16x32) x V (32x16); Vt rows are key-contiguous.
#pragma unroll
        for (int j = 0; j < 4; ++j) {
            const _Float16* vrow = Vbase + (long)(16 * j + la) * S_ + kc + lb * 16;
            Frag bv;
            bv.h[0] = *(const v8h*)(vrow);
            bv.h[1] = *(const v8h*)(vrow + 8);
            O[j] = wmma_f16(ap.v, bv.v, O[j]);
        }
    }

    // Normalize and write f16 output row-major [b][s][h*64+d].
#pragma unroll
    for (int i = 0; i < 8; ++i) {
        const float inv = lrun[i] > 0.f ? 1.f / lrun[i] : 0.f;
        const int srow = q0 + i + lb * 8;
        const long base = ((long)b * S_ + srow) * E_ + (long)hh * D_;
#pragma unroll
        for (int j = 0; j < 4; ++j) {
            Ah[base + j * 16 + la] = (_Float16)(O[j][i] * inv);
        }
    }
}

// ---------------------------------------------------------------------------
// Launch
// ---------------------------------------------------------------------------
extern "C" void kernel_launch(void* const* d_in, const int* in_sizes, int n_in,
                              void* d_out, int out_size, void* d_ws, size_t ws_size,
                              hipStream_t stream) {
    const float* x      = (const float*)d_in[0];
    const int*   mask   = (const int*)d_in[1];
    const float* qkv_w  = (const float*)d_in[2];
    const float* qkv_b  = (const float*)d_in[3];
    const float* proj_w = (const float*)d_in[4];
    const float* proj_b = (const float*)d_in[5];
    float* out = (float*)d_out;

    char* ws = (char*)d_ws;
    const long MB = 1024 * 1024;
    _Float16* xh    = (_Float16*)(ws + 0 * MB);   // 4096*1024 f16 = 8MB
    _Float16* wqkvh = (_Float16*)(ws + 8 * MB);   // 3072*1024 f16 = 6MB
    _Float16* wprjh = (_Float16*)(ws + 14 * MB);  // 1024*1024 f16 = 2MB
    _Float16* Qh    = (_Float16*)(ws + 16 * MB);  // [B,H,S,D] f16 = 8MB
    _Float16* Kh    = (_Float16*)(ws + 24 * MB);  // [B,H,S,D] f16 = 8MB
    _Float16* Vt    = (_Float16*)(ws + 32 * MB);  // [B,H,D,S] f16 = 8MB
    _Float16* Ah    = (_Float16*)(ws + 40 * MB);  // [B,S,E]   f16 = 8MB

    // Stage 0: downconvert activations + weights.
    f32_to_f16_kernel<<<2048, 256, 0, stream>>>(x, xh, (long)B_ * S_ * E_);
    f32_to_f16_kernel<<<2048, 256, 0, stream>>>(qkv_w, wqkvh, (long)3 * E_ * E_);
    f32_to_f16_kernel<<<1024, 256, 0, stream>>>(proj_w, wprjh, (long)E_ * E_);

    // Stage 1: QKV projection (M=4096, N=3072, K=1024); scatter to Q/K/Vt.
    {
        const int wtiles = (B_ * S_ / 32) * (3 * E_ / 64);   // 128*48 = 6144
        gemm_wmma_kernel<<<wtiles / 8, 256, 0, stream>>>(
            xh, wqkvh, qkv_b, B_ * S_, 3 * E_, E_, 0, Qh, Kh, Vt, nullptr);
    }

    // Stage 2: flash attention, 4096 query tiles, 8 waves/block.
    attn_kernel<<<(B_ * H_ * (S_ / 16)) / 8, 256, 0, stream>>>(Qh, Kh, Vt, mask, Ah);

    // Stage 3: output projection (M=4096, N=1024, K=1024) -> f32 out.
    {
        const int wtiles = (B_ * S_ / 32) * (E_ / 64);       // 128*16 = 2048
        gemm_wmma_kernel<<<wtiles / 8, 256, 0, stream>>>(
            Ah, wprjh, proj_b, B_ * S_, E_, E_, 1, nullptr, nullptr, nullptr, out);
    }
}
